// AttentionBlock_72456098284086
// MI455X (gfx1250) — matrix operand order repaired and checked by reading, not compiled, not run on hardware
//
#include <hip/hip_runtime.h>

// -------- problem dims ---------------------------------------------------
constexpr int Bn   = 4;
constexpr int Cdim = 256;
constexpr int Tn   = 64 * 64;        // 4096 tokens
constexpr int NH   = 4;              // heads
constexpr int Dh   = Cdim / NH;      // 64 per head
constexpr int BH   = Bn * NH;        // 16 batched heads
constexpr int NG   = 32;             // groupnorm groups
constexpr float EPSV = 1e-5f;

typedef _Float16 half_t;
typedef _Float16 v16h __attribute__((ext_vector_type(16)));
typedef float    v8f  __attribute__((ext_vector_type(8)));

union HF16 { v16h v; unsigned u[8]; };
union H8v  { uint4 u4; half_t h[8]; };

static __device__ __forceinline__ v8f wmma_f16(v16h a, v16h b, v8f c) {
  // D = A(16x32,f16) * B(32x16,f16) + C(16x16,f32)
  return __builtin_amdgcn_wmma_f32_16x16x32_f16(false, a, false, b, (short)0, c,
                                                false, false);
}

// A-matrix fragment (16xK f16). `row` points at the start of row M=(lane&15).
// K pattern (ISA 7.12.2): lanes0-15 K={0..7,16..23}, lanes16-31 +8.
static __device__ __forceinline__ v16h load_a_frag(const half_t* row, int kBase,
                                                   int lane) {
  HF16 f; const int hi = (lane >= 16) ? 8 : 0;
#pragma unroll
  for (int j = 0; j < 8; ++j) {
    int k = kBase + hi + ((j < 4) ? (2 * j) : (16 + 2 * (j - 4)));
    f.u[j] = *reinterpret_cast<const unsigned*>(row + k);
  }
  return f.v;
}

// B-matrix fragment (32x16 f16) from row-major [n][k] storage; `row` points
// at column n's K-run. lanes0-15 K={0..7,16..23}, lanes16-31 +8.
static __device__ __forceinline__ v16h load_b_frag(const half_t* row, int kBase,
                                                   int lane) {
  HF16 f; const int hi = (lane >= 16) ? 8 : 0;
#pragma unroll
  for (int j = 0; j < 8; ++j) {
    int k = kBase + hi + ((j < 4) ? (2 * j) : (16 + 2 * (j - 4)));
    f.u[j] = *reinterpret_cast<const unsigned*>(row + k);
  }
  return f.v;
}

static __device__ __forceinline__ float half_reduce_max(float v) {
#pragma unroll
  for (int m = 1; m < 16; m <<= 1) v = fmaxf(v, __shfl_xor(v, m, 32));
  return v;
}
static __device__ __forceinline__ float half_reduce_sum(float v) {
#pragma unroll
  for (int m = 1; m < 16; m <<= 1) v += __shfl_xor(v, m, 32);
  return v;
}

// ======================= 1) GroupNorm -> xn[b][t][c] (f16) ================
__global__ __launch_bounds__(256) void gn_kernel(
    const float* __restrict__ x, const float* __restrict__ w,
    const float* __restrict__ gb, half_t* __restrict__ xn) {
  const int b = blockIdx.x / NG, g = blockIdx.x % NG;
  constexpr int CPG = Cdim / NG;                 // 8 channels / group
  const float* xb = x + ((size_t)b * Cdim + g * CPG) * Tn;
  __shared__ float sSum[256], sSq[256];
  float s = 0.f, sq = 0.f;
  const float4* xv = reinterpret_cast<const float4*>(xb);
  for (int i = threadIdx.x; i < (CPG * Tn) / 4; i += 256) {
    float4 v = xv[i];
    s  += v.x + v.y + v.z + v.w;
    sq += v.x * v.x + v.y * v.y + v.z * v.z + v.w * v.w;
  }
  sSum[threadIdx.x] = s; sSq[threadIdx.x] = sq;
  __syncthreads();
  for (int off = 128; off > 0; off >>= 1) {
    if (threadIdx.x < off) {
      sSum[threadIdx.x] += sSum[threadIdx.x + off];
      sSq[threadIdx.x]  += sSq[threadIdx.x + off];
    }
    __syncthreads();
  }
  const float inv_n = 1.f / (CPG * Tn);
  const float mean  = sSum[0] * inv_n;
  const float var   = sSq[0] * inv_n - mean * mean;
  const float rstd  = rsqrtf(var + EPSV);
  half_t* xo = xn + (size_t)b * Tn * Cdim;
  for (int i = threadIdx.x; i < CPG * Tn; i += 256) {
    int cl = i >> 12, t = i & (Tn - 1), c = g * CPG + cl;
    float y = (xb[(size_t)cl * Tn + t] - mean) * rstd * w[c] + gb[c];
    xo[(size_t)t * Cdim + c] = (half_t)y;
  }
}

// ====== 2) QKV GEMM: qkv[o][t] = W[o][:] . xn[:, t] + b ===================
// q -> qT[bh][t][d] (scale folded), k -> kT[bh][t][d], v -> vC[bh][d][t]
__global__ __launch_bounds__(256) void qkv_kernel(
    const float* __restrict__ Wq, const float* __restrict__ bq,
    const half_t* __restrict__ xn, half_t* __restrict__ qT,
    half_t* __restrict__ kT, half_t* __restrict__ vC) {
  const int b = blockIdx.z, o0 = blockIdx.y * 128, t0 = blockIdx.x * 64;
  const int tid = threadIdx.x, wave = tid >> 5, lane = tid & 31;
  const int waveM = wave >> 2, waveN = wave & 3;   // 2 x 4 wave grid
  __shared__ half_t aS[128][32];                   // W tile   [o][k]
  __shared__ half_t bS[64][32];                    // xn tile  [t][k]
  v8f acc[4];
#pragma unroll
  for (int m = 0; m < 4; ++m)
#pragma unroll
    for (int r = 0; r < 8; ++r) acc[m][r] = 0.f;

  for (int kk = 0; kk < Cdim; kk += 32) {
    __syncthreads();
#pragma unroll
    for (int i = 0; i < 4; ++i) {                  // stage W (f32->f16)
      int g = tid + i * 256, row = g >> 3, col = (g & 7) * 4;
      float4 v = *reinterpret_cast<const float4*>(
          Wq + (size_t)(o0 + row) * Cdim + kk + col);
      half_t* d = &aS[row][col];
      d[0] = (half_t)v.x; d[1] = (half_t)v.y;
      d[2] = (half_t)v.z; d[3] = (half_t)v.w;
    }
    {                                              // stage xn tile (b128)
      int row = tid >> 2, seg = (tid & 3) * 8;
      *reinterpret_cast<uint4*>(&bS[row][seg]) =
          *reinterpret_cast<const uint4*>(
              xn + ((size_t)b * Tn + t0 + row) * Cdim + kk + seg);
    }
    __syncthreads();
    v16h bf = load_b_frag(&bS[waveN * 16 + (lane & 15)][0], 0, lane);
#pragma unroll
    for (int m = 0; m < 4; ++m) {
      v16h af = load_a_frag(&aS[waveM * 64 + m * 16 + (lane & 15)][0], 0, lane);
      acc[m] = wmma_f16(af, bf, acc[m]);
    }
  }
  const int t = t0 + waveN * 16 + (lane & 15);
  const int hi8 = (lane >= 16) ? 8 : 0;
#pragma unroll
  for (int m = 0; m < 4; ++m) {
    int ob = o0 + waveM * 64 + m * 16 + hi8;       // 8 consecutive o values
    float vals[8];
#pragma unroll
    for (int r = 0; r < 8; ++r) vals[r] = acc[m][r] + bq[ob + r];
    if (ob < Cdim) {                               // Q (scale 1/sqrt(64))
      H8v p;
#pragma unroll
      for (int r = 0; r < 8; ++r) p.h[r] = (half_t)(vals[r] * 0.125f);
      int c = ob, bh = b * NH + (c >> 6), d0 = c & 63;
      *reinterpret_cast<uint4*>(qT + ((size_t)bh * Tn + t) * Dh + d0) = p.u4;
    } else if (ob < 2 * Cdim) {                    // K
      H8v p;
#pragma unroll
      for (int r = 0; r < 8; ++r) p.h[r] = (half_t)vals[r];
      int c = ob - Cdim, bh = b * NH + (c >> 6), d0 = c & 63;
      *reinterpret_cast<uint4*>(kT + ((size_t)bh * Tn + t) * Dh + d0) = p.u4;
    } else {                                       // V -> [bh][d][t]
      int c = ob - 2 * Cdim, bh = b * NH + (c >> 6), d0 = c & 63;
#pragma unroll
      for (int r = 0; r < 8; ++r)
        vC[((size_t)bh * Dh + d0 + r) * Tn + t] = (half_t)vals[r];
    }
  }
}

// ====== 3) Flash attention: per wave 32 query rows (2 tiles), online ======
// K/V B-fragments are shared across the two query tiles -> 2 WMMAs per
// LDS fragment fetch; block barriers amortized over 256 query rows.
__global__ __launch_bounds__(256) void attn_kernel(
    const half_t* __restrict__ qT, const half_t* __restrict__ kT,
    const half_t* __restrict__ vC, half_t* __restrict__ aOut) {
  const int bh = blockIdx.y, tBlk = blockIdx.x * 256;
  const int tid = threadIdx.x, wave = tid >> 5, lane = tid & 31;
  const int t0 = tBlk + wave * 32;
  const int b = bh / NH, head = bh % NH;
  const int hi8 = (lane >= 16) ? 8 : 0;
  __shared__ half_t kS[32][Dh];         // [s][d]
  __shared__ half_t vS[Dh][32];         // [d][s]
  __shared__ half_t pS[8][2][16][32];   // per-wave P scratch (C->A relayout)

  v16h aq[2][2];                        // [query tile][d chunk]
#pragma unroll
  for (int qt = 0; qt < 2; ++qt) {
    const half_t* qrow =
        qT + ((size_t)bh * Tn + t0 + qt * 16 + (lane & 15)) * Dh;
    aq[qt][0] = load_a_frag(qrow, 0, lane);
    aq[qt][1] = load_a_frag(qrow, 32, lane);
  }

  v8f accO[2][4]; float mrow[2][8], lrow[2][8];
#pragma unroll
  for (int qt = 0; qt < 2; ++qt) {
#pragma unroll
    for (int dn = 0; dn < 4; ++dn)
#pragma unroll
      for (int r = 0; r < 8; ++r) accO[qt][dn][r] = 0.f;
#pragma unroll
    for (int r = 0; r < 8; ++r) { mrow[qt][r] = -1e30f; lrow[qt][r] = 0.f; }
  }

  for (int s0 = 0; s0 < Tn; s0 += 32) {
    __syncthreads();
    {                                              // stage K block (b128)
      int row = tid >> 3, seg = (tid & 7) * 8;
      const half_t* src = kT + ((size_t)bh * Tn + s0 + row) * Dh + seg;
      *reinterpret_cast<uint4*>(&kS[row][seg]) =
          *reinterpret_cast<const uint4*>(src);
      if (s0 + 32 < Tn) __builtin_prefetch(src + 32 * Dh, 0, 1);
    }
#pragma unroll
    for (int i = 0; i < 2; ++i) {                  // stage V block (b64)
      int e = tid + i * 256, d = e >> 3, seg = (e & 7) * 4;
      const half_t* src = vC + ((size_t)bh * Dh + d) * Tn + s0 + seg;
      *reinterpret_cast<uint2*>(&vS[d][seg]) =
          *reinterpret_cast<const uint2*>(src);
      if (s0 + 32 < Tn) __builtin_prefetch(src + 32, 0, 1);
    }
    __syncthreads();

    // ---- S = (Q^T K): 2 query tiles x 2 key tiles, shared B frags ----
    v8f sAcc[2][2];
#pragma unroll
    for (int sn = 0; sn < 2; ++sn) {
      const half_t* krow = &kS[sn * 16 + (lane & 15)][0];
      v16h bk0 = load_b_frag(krow, 0, lane);       // d 0..31
      v16h bk1 = load_b_frag(krow, 32, lane);      // d 32..63
#pragma unroll
      for (int qt = 0; qt < 2; ++qt) {
        v8f z;
#pragma unroll
        for (int r = 0; r < 8; ++r) z[r] = 0.f;
        z = wmma_f16(aq[qt][0], bk0, z);
        z = wmma_f16(aq[qt][1], bk1, z);
        sAcc[qt][sn] = z;
      }
    }
    // ---- online softmax (rows live across 16-lane half-waves) ----
#pragma unroll
    for (int qt = 0; qt < 2; ++qt) {
#pragma unroll
      for (int r = 0; r < 8; ++r) {
        float v0 = sAcc[qt][0][r], v1 = sAcc[qt][1][r];
        float mx = half_reduce_max(fmaxf(v0, v1));
        float mnew = fmaxf(mrow[qt][r], mx);
        float alpha = __expf(mrow[qt][r] - mnew);
        float p0 = __expf(v0 - mnew), p1 = __expf(v1 - mnew);
        lrow[qt][r] = lrow[qt][r] * alpha + half_reduce_sum(p0 + p1);
        mrow[qt][r] = mnew;
#pragma unroll
        for (int dn = 0; dn < 4; ++dn) accO[qt][dn][r] *= alpha;
        pS[wave][qt][r + hi8][lane & 15]        = (half_t)p0;
        pS[wave][qt][r + hi8][16 + (lane & 15)] = (half_t)p1;
      }
    }
    v16h ap0 = load_a_frag(&pS[wave][0][lane & 15][0], 0, lane);
    v16h ap1 = load_a_frag(&pS[wave][1][lane & 15][0], 0, lane);
#pragma unroll
    for (int dn = 0; dn < 4; ++dn) {               // O += P * V^T
      v16h bv = load_b_frag(&vS[dn * 16 + (lane & 15)][0], 0, lane);
      accO[0][dn] = wmma_f16(ap0, bv, accO[0][dn]);
      accO[1][dn] = wmma_f16(ap1, bv, accO[1][dn]);
    }
  }
#pragma unroll
  for (int qt = 0; qt < 2; ++qt) {                 // write a[b][t][c]
#pragma unroll
    for (int r = 0; r < 8; ++r) {
      float inv = 1.f / lrow[qt][r];
      int t = t0 + qt * 16 + r + hi8;
#pragma unroll
      for (int dn = 0; dn < 4; ++dn) {
        int c = head * Dh + dn * 16 + (lane & 15);
        aOut[((size_t)b * Tn + t) * Cdim + c] = (half_t)(accO[qt][dn][r] * inv);
      }
    }
  }
}

// ====== 4) Proj GEMM + bias + residual (f32 out) ==========================
__global__ __launch_bounds__(256) void proj_kernel(
    const float* __restrict__ Wp, const float* __restrict__ bp,
    const half_t* __restrict__ aIn, const float* __restrict__ x,
    float* __restrict__ out) {
  const int b = blockIdx.z, o0 = blockIdx.y * 128, t0 = blockIdx.x * 64;
  const int tid = threadIdx.x, wave = tid >> 5, lane = tid & 31;
  const int waveM = wave >> 2, waveN = wave & 3;
  __shared__ half_t aS[128][32];
  __shared__ half_t bS[64][32];
  v8f acc[4];
#pragma unroll
  for (int m = 0; m < 4; ++m)
#pragma unroll
    for (int r = 0; r < 8; ++r) acc[m][r] = 0.f;

  for (int kk = 0; kk < Cdim; kk += 32) {
    __syncthreads();
#pragma unroll
    for (int i = 0; i < 4; ++i) {
      int g = tid + i * 256, row = g >> 3, col = (g & 7) * 4;
      float4 v = *reinterpret_cast<const float4*>(
          Wp + (size_t)(o0 + row) * Cdim + kk + col);
      half_t* d = &aS[row][col];
      d[0] = (half_t)v.x; d[1] = (half_t)v.y;
      d[2] = (half_t)v.z; d[3] = (half_t)v.w;
    }
    {
      int row = tid >> 2, seg = (tid & 3) * 8;
      *reinterpret_cast<uint4*>(&bS[row][seg]) =
          *reinterpret_cast<const uint4*>(
              aIn + ((size_t)b * Tn + t0 + row) * Cdim + kk + seg);
    }
    __syncthreads();
    v16h bf = load_b_frag(&bS[waveN * 16 + (lane & 15)][0], 0, lane);
#pragma unroll
    for (int m = 0; m < 4; ++m) {
      v16h af = load_a_frag(&aS[waveM * 64 + m * 16 + (lane & 15)][0], 0, lane);
      acc[m] = wmma_f16(af, bf, acc[m]);
    }
  }
  const int t = t0 + waveN * 16 + (lane & 15);
  const int hi8 = (lane >= 16) ? 8 : 0;
#pragma unroll
  for (int m = 0; m < 4; ++m) {
    int ob = o0 + waveM * 64 + m * 16 + hi8;
#pragma unroll
    for (int r = 0; r < 8; ++r) {
      int o = ob + r;
      size_t idx = ((size_t)b * Cdim + o) * Tn + t;
      out[idx] = acc[m][r] + bp[o] + x[idx];       // + residual
    }
  }
}

// ==========================================================================
extern "C" void kernel_launch(void* const* d_in, const int* in_sizes, int n_in,
                              void* d_out, int out_size, void* d_ws,
                              size_t ws_size, hipStream_t stream) {
  const float* x     = (const float*)d_in[0];
  const float* gnw   = (const float*)d_in[1];
  const float* gnb   = (const float*)d_in[2];
  const float* qkvw  = (const float*)d_in[3];
  const float* qkvb  = (const float*)d_in[4];
  const float* projw = (const float*)d_in[5];
  const float* projb = (const float*)d_in[6];
  float* out = (float*)d_out;

  // workspace: 5 f16 tensors of 4M halves each = 40 MiB total
  half_t* ws = (half_t*)d_ws;
  half_t* xn = ws;                                 // [b][t][c]
  half_t* qT = xn + (size_t)Bn * Tn * Cdim;        // [bh][t][d] (pre-scaled)
  half_t* kT = qT + (size_t)BH * Tn * Dh;          // [bh][t][d]
  half_t* vC = kT + (size_t)BH * Tn * Dh;          // [bh][d][t]
  half_t* aO = vC + (size_t)BH * Tn * Dh;          // [b][t][c]

  gn_kernel<<<Bn * NG, 256, 0, stream>>>(x, gnw, gnb, xn);
  qkv_kernel<<<dim3(Tn / 64, (3 * Cdim) / 128, Bn), 256, 0, stream>>>(
      qkvw, qkvb, xn, qT, kT, vC);
  attn_kernel<<<dim3(Tn / 256, BH), 256, 0, stream>>>(qT, kT, vC, aO);
  proj_kernel<<<dim3(Tn / 64, Cdim / 128, Bn), 256, 0, stream>>>(
      projw, projb, aO, x, out);
}
